// ScaledDotProductAttention_58755152609364
// MI455X (gfx1250) — matrix operand order repaired and checked by reading, not compiled, not run on hardware
//
#include <hip/hip_runtime.h>

// B=4, H=16, L=2048, D=64; out = softmax((q/8) @ k^T) @ v, plus attn matrix.
constexpr int Bb = 4, Hh = 16, Ll = 2048, Dd = 64;
#define TEMP_INV 0.125f

typedef __attribute__((ext_vector_type(16))) __bf16 v16bf;
typedef __attribute__((ext_vector_type(8)))  __bf16 v8bf;
typedef __attribute__((ext_vector_type(8)))  float  v8f;
typedef __attribute__((ext_vector_type(8)))  unsigned v8u;

// f32 -> bf16, round-half-up (bias + take high half)
__device__ inline unsigned short f2bf(float f) {
    return (unsigned short)((__builtin_bit_cast(unsigned, f) + 0x8000u) >> 16);
}
// packed f32x2 -> bf16x2: two biased adds + one v_perm_b32 (bytes {b3,b2,a3,a2})
__device__ inline unsigned pk2bf(float a, float b) {
    unsigned au = __builtin_bit_cast(unsigned, a) + 0x8000u;
    unsigned bu = __builtin_bit_cast(unsigned, b) + 0x8000u;
    return __builtin_amdgcn_perm(bu, au, 0x07060302u);
}
__device__ inline v16bf pack16(const float* f) {
    v8u u;
#pragma unroll
    for (int i = 0; i < 8; ++i) u[i] = pk2bf(f[2 * i], f[2 * i + 1]);
    return __builtin_bit_cast(v16bf, u);
}
__device__ inline v16bf cat8(v8bf a, v8bf b) {
    return __builtin_shufflevector(a, b, 0,1,2,3,4,5,6,7,8,9,10,11,12,13,14,15);
}
__device__ inline v8f wmma_bf16(v16bf a, v16bf b, v8f c) {
    return __builtin_amdgcn_wmma_f32_16x16x32_bf16(
        false, a, false, b, (short)0, c, false, false);
}
// LDS (16B per lane) -> global, async DMA path; tracked by ASYNCcnt.
__device__ inline void async_store_b128(float* gaddr, const float* lds_generic) {
    unsigned loff = (unsigned)(unsigned long long)
        (const __attribute__((address_space(3))) char*)lds_generic;
    asm volatile("global_store_async_from_lds_b128 %0, %1, off"
                 :: "v"((unsigned long long)gaddr), "v"(loff) : "memory");
}

__global__ __launch_bounds__(256, 2)
void fattn_kernel(const float* __restrict__ q, const float* __restrict__ k,
                  const float* __restrict__ v, float* __restrict__ out,
                  float* __restrict__ attn) {
    __shared__ __align__(16) unsigned short Kb[32 * 64];    // K tile, bf16, [key][d]
    __shared__ __align__(16) unsigned short Vt[64 * 32];    // V^T tile, bf16, [d][key]
    __shared__ __align__(16) float          Ps[8 * 16 * 32]; // per-wave P, [wave][q][key]

    const int tid  = threadIdx.x;
    const int wv   = tid >> 5;
    const int lane = tid & 31;
    const int m    = lane & 15;   // query column in S^T C-layout
    const int h    = lane >> 4;   // half-wave select
    const int bh   = blockIdx.y;
    const int q0w  = blockIdx.x * 128 + wv * 16;
    const int qrow = q0w + m;

    const float* qb = q + (size_t)bh * Ll * Dd;
    const float* kb = k + (size_t)bh * Ll * Dd;
    const float* vb = v + (size_t)bh * Ll * Dd;
    float* ob = out  + (size_t)bh * Ll * Dd;
    float* ab = attn + (size_t)bh * Ll * Ll;

    // ---- Q B-fragments (held for the whole kernel). B layout: elem i -> d = dbase + 16*(i>>3) + 8*h + (i&7)
    v16bf bq[2];
#pragma unroll
    for (int db = 0; db < 2; ++db) {
        const float4* s4 = (const float4*)(qb + (size_t)qrow * Dd + db * 32 + 8 * h);
        float4 c0 = s4[0], c1 = s4[1], c2 = s4[4], c3 = s4[5];
        float f[16] = { c0.x * TEMP_INV, c0.y * TEMP_INV, c0.z * TEMP_INV, c0.w * TEMP_INV,
                        c1.x * TEMP_INV, c1.y * TEMP_INV, c1.z * TEMP_INV, c1.w * TEMP_INV,
                        c2.x * TEMP_INV, c2.y * TEMP_INV, c2.z * TEMP_INV, c2.w * TEMP_INV,
                        c3.x * TEMP_INV, c3.y * TEMP_INV, c3.z * TEMP_INV, c3.w * TEMP_INV };
        bq[db] = pack16(f);
    }

    auto stageK = [&](int kk) {
        int row = tid >> 3, dc = (tid & 7) * 8;
        const float4* s4 = (const float4*)(kb + (size_t)(kk + row) * Dd + dc);
        float4 a = s4[0], b2 = s4[1];
        uint4 p;
        p.x = pk2bf(a.x,  a.y);
        p.y = pk2bf(a.z,  a.w);
        p.z = pk2bf(b2.x, b2.y);
        p.w = pk2bf(b2.z, b2.w);
        *(uint4*)&Kb[row * 64 + dc] = p;
    };
    auto stageV = [&](int kk) {
        int row = tid >> 3, dc = (tid & 7) * 8;
        const float4* s4 = (const float4*)(vb + (size_t)(kk + row) * Dd + dc);
        float4 a = s4[0], b2 = s4[1];
        float f8[8] = { a.x, a.y, a.z, a.w, b2.x, b2.y, b2.z, b2.w };
#pragma unroll
        for (int i = 0; i < 8; ++i) Vt[(dc + i) * 32 + row] = f2bf(f8[i]);
    };

    // S^T tiles: cst[t] covers keys [16t,16t+16) x queries [q0w,q0w+16).
    // A (K-tile) layout: elem i -> d = dbase + 16*(i>>3) + 8*h + (i&7)  (two b128 chunks)
    auto scoreT = [&](v8f cst[2]) {
#pragma unroll
        for (int t = 0; t < 2; ++t) {
            v8f c = {};
#pragma unroll
            for (int db = 0; db < 2; ++db) {
                int rl = 16 * t + m;
                const v8bf* p0 = (const v8bf*)&Kb[rl * 64 + db * 32 + 8 * h];
                const v8bf* p1 = (const v8bf*)&Kb[rl * 64 + db * 32 + 16 + 8 * h];
                c = wmma_bf16(cat8(*p0, *p1), bq[db], c);
            }
            cst[t] = c;
        }
    };

    // ---------------- Pass 1: row max + sum(exp) (streaming, no stores) ----------------
    float runM = -1e30f, runL = 0.f;
    for (int blk = 0; blk < Ll / 32; ++blk) {
        __syncthreads();
        stageK(blk * 32);
        if (blk + 1 < Ll / 32) {
            int row = tid >> 3, dc = (tid & 7) * 8;
            __builtin_prefetch(kb + (size_t)((blk + 1) * 32 + row) * Dd + dc, 0, 1);
        }
        __syncthreads();
        v8f cst[2];
        scoreT(cst);
#pragma unroll
        for (int t = 0; t < 2; ++t) {
            float tm = cst[t][0];
#pragma unroll
            for (int i = 1; i < 8; ++i) tm = fmaxf(tm, cst[t][i]);
            float nm = fmaxf(runM, tm);
            float s = 0.f;
#pragma unroll
            for (int i = 0; i < 8; ++i) s += __expf(cst[t][i] - nm);
            runL = runL * __expf(runM - nm) + s;
            runM = nm;
        }
    }
    {   // combine the two half-wave partial stats (lane l <-> l^16 hold same query row)
        float om = __shfl_xor(runM, 16);
        float ol = __shfl_xor(runL, 16);
        float M  = fmaxf(runM, om);
        runL = runL * __expf(runM - M) + ol * __expf(om - M);
        runM = M;
    }
    const float Mrow = runM;
    const float invL = 1.f / runL;

    // ---------------- Pass 2: P write (async LDS->global) + O^T = V^T @ P^T ----------------
    v8f oacc[4] = {};
    for (int blk = 0; blk < Ll / 32; ++blk) {
        __syncthreads();
        stageK(blk * 32);
        stageV(blk * 32);
        if (blk + 1 < Ll / 32) {
            int row = tid >> 3, dc = (tid & 7) * 8;
            __builtin_prefetch(kb + (size_t)((blk + 1) * 32 + row) * Dd + dc, 0, 1);
            __builtin_prefetch(vb + (size_t)((blk + 1) * 32 + row) * Dd + dc, 0, 1);
        }
        __syncthreads();
        v8f cst[2];
        scoreT(cst);

        // previous block's async attn stores must be done before Ps is overwritten
        asm volatile("s_wait_asynccnt 0x0" ::: "memory");

        // normalized probabilities -> per-wave LDS staging buffer Ps[wv][q][key]
#pragma unroll
        for (int t = 0; t < 2; ++t) {
            float pv[8];
#pragma unroll
            for (int i = 0; i < 8; ++i) pv[i] = __expf(cst[t][i] - Mrow) * invL;
            float4* dst = (float4*)&Ps[(wv * 16 + m) * 32 + 16 * t + 8 * h];
            dst[0] = make_float4(pv[0], pv[1], pv[2], pv[3]);
            dst[1] = make_float4(pv[4], pv[5], pv[6], pv[7]);
        }
        // make the DS writes visible to the async DMA engine, then stream the
        // attn tile straight LDS -> global (512B contiguous per instruction)
        asm volatile("s_wait_dscnt 0x0" ::: "memory");
#pragma unroll
        for (int i = 0; i < 4; ++i) {
            int qq   = i * 4 + (lane >> 3);
            int kloc = (lane & 7) * 4;
            async_store_b128(&ab[(size_t)(q0w + qq) * Ll + blk * 32 + kloc],
                             &Ps[wv * 512 + i * 128 + lane * 4]);
        }
        // P^T B-fragment (elem i -> key = 16*(i>>3) + 8*h + (i&7)) reloaded from Ps (handles half swap)
        v16bf bp;
        {
            const float4* s4 = (const float4*)&Ps[(wv * 16 + m) * 32 + 8 * h];
            float4 c0 = s4[0], c1 = s4[1], c2 = s4[4], c3 = s4[5];
            float f[16] = { c0.x,c0.y,c0.z,c0.w, c1.x,c1.y,c1.z,c1.w,
                            c2.x,c2.y,c2.z,c2.w, c3.x,c3.y,c3.z,c3.w };
            bp = pack16(f);
        }
        // O^T accumulate: A = V^T (16 d-rows x 32 keys), 4 row tiles cover D=64
#pragma unroll
        for (int mt = 0; mt < 4; ++mt) {
            int dr = 16 * mt + m;
            const v8bf* p0 = (const v8bf*)&Vt[dr * 32 + 8 * h];
            const v8bf* p1 = (const v8bf*)&Vt[dr * 32 + 16 + 8 * h];
            oacc[mt] = wmma_bf16(cat8(*p0, *p1), bp, oacc[mt]);
        }
    }
    asm volatile("s_wait_asynccnt 0x0" ::: "memory");

    // ---- store O: lane holds query qrow, d = 16*mt + 8*h + v (contiguous per v)
#pragma unroll
    for (int mt = 0; mt < 4; ++mt) {
        int d0 = 16 * mt + 8 * h;
        float4* dst = (float4*)&ob[(size_t)qrow * Dd + d0];
        dst[0] = make_float4(oacc[mt][0], oacc[mt][1], oacc[mt][2], oacc[mt][3]);
        dst[1] = make_float4(oacc[mt][4], oacc[mt][5], oacc[mt][6], oacc[mt][7]);
    }
}

extern "C" void kernel_launch(void* const* d_in, const int* in_sizes, int n_in,
                              void* d_out, int out_size, void* d_ws, size_t ws_size,
                              hipStream_t stream) {
    (void)in_sizes; (void)n_in; (void)out_size; (void)d_ws; (void)ws_size;
    const float* q = (const float*)d_in[0];
    const float* k = (const float*)d_in[1];
    const float* v = (const float*)d_in[2];
    float* out  = (float*)d_out;
    float* attn = out + (size_t)Bb * Hh * Ll * Dd;   // tuple: (out, attn) concatenated
    dim3 grid(Ll / 128, Bb * Hh);
    fattn_kernel<<<grid, 256, 0, stream>>>(q, k, v, out, attn);
}
